// GroupingLayer_19670950216060
// MI455X (gfx1250) — matrix operand order repaired and checked by reading, not compiled, not run on hardware
//
#include <hip/hip_runtime.h>
#include <hip/hip_bf16.h>
#include <math.h>

// ---------------- problem constants (GroupViT GroupingLayer) ----------------
#define B_     16
#define NTOK   784
#define GTOK   64
#define SEQ    848          // NTOK + GTOK
#define CH     768
#define FF     3072
#define HEADS_ 12
#define HD     64
#define DEPTH_ 6
#define MROWS  (B_*SEQ)     // 13568 = 106*128

typedef __bf16 bf16_t;
typedef bf16_t v16bf __attribute__((ext_vector_type(16)));
typedef float  v8f   __attribute__((ext_vector_type(8)));

union Frag {
    v16bf v;
    unsigned short u[16];
    uint4 q[2];
};

__device__ __forceinline__ unsigned short f2bf(float x) {
    unsigned int u = __float_as_uint(x);
    u += 0x7FFFu + ((u >> 16) & 1u);       // round-to-nearest-even
    return (unsigned short)(u >> 16);
}

__device__ __forceinline__ float gelu_exact(float x) {
    return 0.5f * x * (1.f + erff(x * 0.70710678118654752f));
}

// ---------------------------------------------------------------------------
// fp32 -> bf16 weight pre-conversion (hoists all conversion VALU out of GEMMs)
// ---------------------------------------------------------------------------
__global__ void f32_to_bf16_kernel(const float* __restrict__ in,
                                   unsigned short* __restrict__ out, int n)
{
    const int i = blockIdx.x * 256 + threadIdx.x;
    if (i < n) out[i] = f2bf(in[i]);
}

// ---------------------------------------------------------------------------
// WMMA GEMM: out[M,N] = A[M,K](bf16) * Wb[N,K]^T(bf16) + bias
// One wave computes a 16x64 strip (A fragment reused across 4 WMMAs).
// Block = 8 waves -> 128x64 tile. Optional residual (fp32) / exact GELU.
// ---------------------------------------------------------------------------
__global__ void gemm_wmma(const unsigned short* __restrict__ A,
                          const unsigned short* __restrict__ Wb,
                          const float* __restrict__ bias,
                          const float* __restrict__ resid,
                          float* __restrict__ outF,
                          unsigned short* __restrict__ outH,
                          int M, int N, int K, int act)
{
    const int lane = threadIdx.x & 31;
    const int wave = threadIdx.x >> 5;
    const int m0 = blockIdx.y * 128 + wave * 16;
    const int n0 = blockIdx.x * 64;
    if (m0 >= M) return;
    const int li = lane & 15, hi = lane >> 4;

    const unsigned short* arow = A + (size_t)(m0 + li) * K;
    const unsigned short* w0 = Wb + (size_t)(n0 + 0  + li) * K;
    const unsigned short* w1 = Wb + (size_t)(n0 + 16 + li) * K;
    const unsigned short* w2 = Wb + (size_t)(n0 + 32 + li) * K;
    const unsigned short* w3 = Wb + (size_t)(n0 + 48 + li) * K;

    v8f c0 = {}, c1 = {}, c2 = {}, c3 = {};
    for (int k0 = 0; k0 < K; k0 += 32) {
        __builtin_prefetch((const void*)(arow + k0 + 128), 0, 0);
        Frag a, b0, b1, b2, b3;
        // A 16x32 fragment: row = li, K = k0 + hi*16 + j (16 contiguous bf16)
        a.q[0] = *(const uint4*)(arow + k0 + hi * 16);
        a.q[1] = *(const uint4*)(arow + k0 + hi * 16 + 8);
        // B 32x16 fragments: col = li, K = k0 + hi*16 + j (16 contiguous bf16)
        b0.q[0] = *(const uint4*)(w0 + k0 + hi * 16);
        b0.q[1] = *(const uint4*)(w0 + k0 + hi * 16 + 8);
        b1.q[0] = *(const uint4*)(w1 + k0 + hi * 16);
        b1.q[1] = *(const uint4*)(w1 + k0 + hi * 16 + 8);
        b2.q[0] = *(const uint4*)(w2 + k0 + hi * 16);
        b2.q[1] = *(const uint4*)(w2 + k0 + hi * 16 + 8);
        b3.q[0] = *(const uint4*)(w3 + k0 + hi * 16);
        b3.q[1] = *(const uint4*)(w3 + k0 + hi * 16 + 8);
        c0 = __builtin_amdgcn_wmma_f32_16x16x32_bf16(false, a.v, false, b0.v, (short)0, c0, false, false);
        c1 = __builtin_amdgcn_wmma_f32_16x16x32_bf16(false, a.v, false, b1.v, (short)0, c1, false, false);
        c2 = __builtin_amdgcn_wmma_f32_16x16x32_bf16(false, a.v, false, b2.v, (short)0, c2, false, false);
        c3 = __builtin_amdgcn_wmma_f32_16x16x32_bf16(false, a.v, false, b3.v, (short)0, c3, false, false);
    }

    #pragma unroll
    for (int t = 0; t < 4; ++t) {
        const v8f& c = (t == 0) ? c0 : (t == 1) ? c1 : (t == 2) ? c2 : c3;
        const int col = n0 + t * 16 + li;
        const float bval = bias[col];
        #pragma unroll
        for (int i = 0; i < 8; ++i) {
            const int row = m0 + i + hi * 8;
            float v = c[i] + bval;
            if (act == 1) v = gelu_exact(v);
            const size_t idx = (size_t)row * N + col;
            if (resid) v += resid[idx];
            if (outF) outF[idx] = v;
            if (outH) outH[idx] = f2bf(v);
        }
    }
}

// ---------------------------------------------------------------------------
// LayerNorm (fp32 in, bf16 out), one 256-thread block per token row.
// ---------------------------------------------------------------------------
__global__ void layernorm_kernel(const float* __restrict__ x,
                                 const float* __restrict__ g,
                                 const float* __restrict__ b,
                                 unsigned short* __restrict__ out)
{
    __shared__ float red[256];
    const int row = blockIdx.x, tid = threadIdx.x;
    const float* xr = x + (size_t)row * CH;

    float s = 0.f;
    for (int c = tid; c < CH; c += 256) s += xr[c];
    red[tid] = s; __syncthreads();
    for (int off = 128; off > 0; off >>= 1) { if (tid < off) red[tid] += red[tid + off]; __syncthreads(); }
    const float mean = red[0] / (float)CH;
    __syncthreads();

    float vs = 0.f;
    for (int c = tid; c < CH; c += 256) { float d = xr[c] - mean; vs += d * d; }
    red[tid] = vs; __syncthreads();
    for (int off = 128; off > 0; off >>= 1) { if (tid < off) red[tid] += red[tid + off]; __syncthreads(); }
    const float rstd = rsqrtf(red[0] / (float)CH + 1e-5f);

    for (int c = tid; c < CH; c += 256)
        out[(size_t)row * CH + c] = f2bf((xr[c] - mean) * rstd * g[c] + b[c]);
}

// ---------------------------------------------------------------------------
// V transpose: vt[b, c, s] = qkv[b, s, 2C + c]  (bf16), enables contiguous
// 16B B-fragment loads in the PV WMMA.
// ---------------------------------------------------------------------------
__global__ void transpose_v_kernel(const unsigned short* __restrict__ qkv,
                                   unsigned short* __restrict__ vt)
{
    const int idx = blockIdx.x * 256 + threadIdx.x;
    if (idx >= B_ * SEQ * CH) return;
    const int c = idx % CH;
    const int s = (idx / CH) % SEQ;
    const int b = idx / (CH * SEQ);
    vt[((size_t)b * CH + c) * SEQ + s] =
        qkv[(size_t)(b * SEQ + s) * (3 * CH) + 2 * CH + c];
}

// ---------------------------------------------------------------------------
// Flash-attention: one wave per (b, h, q-tile of 16 rows), hd = 64.
// Online softmax over 27 kv-chunks of 32; QK^T = 4 WMMA, PV = 4 WMMA per chunk.
// P routed to A-fragment layout through 1KB of LDS (s_wait_dscnt 0).
// ---------------------------------------------------------------------------
__global__ void attn_kernel(const unsigned short* __restrict__ qkv,  // bf16 [B,S,3C]
                            const unsigned short* __restrict__ vt,   // bf16 [B,C,S]
                            unsigned short* __restrict__ obuf)       // bf16 [B,S,C]
{
    const int C3 = 3 * CH;
    const int lane = threadIdx.x;
    const int li = lane & 15, hi = lane >> 4;
    const int qt = blockIdx.x, h = blockIdx.y, b = blockIdx.z;
    const float scale = 0.125f;   // hd^-0.5 = 1/8

    __shared__ __align__(16) unsigned short pl[16 * 32];

    // Q fragments for both 32-wide d-chunks (resident for whole kernel)
    const unsigned short* qrow = qkv + (size_t)(b * SEQ + qt * 16 + li) * C3 + h * HD;
    Frag qa0, qa1;
    qa0.q[0] = *(const uint4*)(qrow + hi * 16);
    qa0.q[1] = *(const uint4*)(qrow + hi * 16 + 8);
    qa1.q[0] = *(const uint4*)(qrow + 32 + hi * 16);
    qa1.q[1] = *(const uint4*)(qrow + 32 + hi * 16 + 8);

    // V rows (transposed layout): one row per output d-column tile
    const unsigned short* v0 = vt + ((size_t)b * CH + h * HD + 0  + li) * SEQ;
    const unsigned short* v1 = vt + ((size_t)b * CH + h * HD + 16 + li) * SEQ;
    const unsigned short* v2 = vt + ((size_t)b * CH + h * HD + 32 + li) * SEQ;
    const unsigned short* v3 = vt + ((size_t)b * CH + h * HD + 48 + li) * SEQ;

    v8f acc0 = {}, acc1 = {}, acc2 = {}, acc3 = {};
    float mrow[8], lrow[8];
    #pragma unroll
    for (int i = 0; i < 8; ++i) { mrow[i] = -3.0e38f; lrow[i] = 0.f; }

    const int nchunk = (SEQ + 31) / 32;   // 27 (last chunk has 16 valid tokens)
    for (int kc = 0; kc < nchunk; ++kc) {
        const int kv0 = kc * 32;
        float p[2][8];

        #pragma unroll
        for (int t = 0; t < 2; ++t) {
            const int s0 = kv0 + t * 16;
            if (s0 < SEQ) {   // uniform per wave
                const unsigned short* krow =
                    qkv + (size_t)(b * SEQ + s0 + li) * C3 + CH + h * HD;
                Frag kb0, kb1;
                kb0.q[0] = *(const uint4*)(krow + hi * 16);
                kb0.q[1] = *(const uint4*)(krow + hi * 16 + 8);
                kb1.q[0] = *(const uint4*)(krow + 32 + hi * 16);
                kb1.q[1] = *(const uint4*)(krow + 32 + hi * 16 + 8);
                v8f cc = {};
                cc = __builtin_amdgcn_wmma_f32_16x16x32_bf16(false, qa0.v, false, kb0.v,
                                                             (short)0, cc, false, false);
                cc = __builtin_amdgcn_wmma_f32_16x16x32_bf16(false, qa1.v, false, kb1.v,
                                                             (short)0, cc, false, false);
                #pragma unroll
                for (int i = 0; i < 8; ++i) p[t][i] = cc[i] * scale;
            } else {
                #pragma unroll
                for (int i = 0; i < 8; ++i) p[t][i] = -3.0e38f;
            }
        }

        // online softmax update (row r = i + hi*8; its 16 cols live across one 16-lane half)
        #pragma unroll
        for (int i = 0; i < 8; ++i) {
            float v = fmaxf(p[0][i], p[1][i]);
            v = fmaxf(v, __shfl_xor(v, 1));
            v = fmaxf(v, __shfl_xor(v, 2));
            v = fmaxf(v, __shfl_xor(v, 4));
            v = fmaxf(v, __shfl_xor(v, 8));
            const float mn = fmaxf(mrow[i], v);
            const float corr = __expf(mrow[i] - mn);
            mrow[i] = mn;
            const float p0 = __expf(p[0][i] - mn);
            const float p1 = __expf(p[1][i] - mn);
            p[0][i] = p0; p[1][i] = p1;
            float rs = p0 + p1;
            rs += __shfl_xor(rs, 1);
            rs += __shfl_xor(rs, 2);
            rs += __shfl_xor(rs, 4);
            rs += __shfl_xor(rs, 8);
            lrow[i] = lrow[i] * corr + rs;
            acc0[i] *= corr; acc1[i] *= corr; acc2[i] *= corr; acc3[i] *= corr;
        }

        // stage P (16x32 bf16, row-major) to LDS: C-layout -> A-fragment layout
        #pragma unroll
        for (int i = 0; i < 8; ++i) {
            pl[(i + hi * 8) * 32 + li]      = f2bf(p[0][i]);
            pl[(i + hi * 8) * 32 + 16 + li] = f2bf(p[1][i]);
        }
        asm volatile("s_wait_dscnt 0" ::: "memory");

        Frag pa;
        const unsigned short* pr = pl + li * 32;
        pa.q[0] = *(const uint4*)(pr + hi * 16);
        pa.q[1] = *(const uint4*)(pr + hi * 16 + 8);

        // PV: contiguous B-fragment loads from transposed V (masked tail has P=0)
        Frag vb0, vb1, vb2, vb3;
        vb0.q[0] = *(const uint4*)(v0 + kv0 + hi * 16);
        vb0.q[1] = *(const uint4*)(v0 + kv0 + hi * 16 + 8);
        vb1.q[0] = *(const uint4*)(v1 + kv0 + hi * 16);
        vb1.q[1] = *(const uint4*)(v1 + kv0 + hi * 16 + 8);
        vb2.q[0] = *(const uint4*)(v2 + kv0 + hi * 16);
        vb2.q[1] = *(const uint4*)(v2 + kv0 + hi * 16 + 8);
        vb3.q[0] = *(const uint4*)(v3 + kv0 + hi * 16);
        vb3.q[1] = *(const uint4*)(v3 + kv0 + hi * 16 + 8);
        acc0 = __builtin_amdgcn_wmma_f32_16x16x32_bf16(false, pa.v, false, vb0.v, (short)0, acc0, false, false);
        acc1 = __builtin_amdgcn_wmma_f32_16x16x32_bf16(false, pa.v, false, vb1.v, (short)0, acc1, false, false);
        acc2 = __builtin_amdgcn_wmma_f32_16x16x32_bf16(false, pa.v, false, vb2.v, (short)0, acc2, false, false);
        acc3 = __builtin_amdgcn_wmma_f32_16x16x32_bf16(false, pa.v, false, vb3.v, (short)0, acc3, false, false);
        asm volatile("" ::: "memory");   // keep next iter's LDS stores after these loads
    }

    // O = acc / l  ->  bf16 [B,S,C] at column block h*HD
    #pragma unroll
    for (int i = 0; i < 8; ++i) {
        const int row = qt * 16 + i + hi * 8;
        const float inv = 1.f / lrow[i];
        const size_t base = (size_t)(b * SEQ + row) * CH + h * HD + li;
        obuf[base +  0] = f2bf(acc0[i] * inv);
        obuf[base + 16] = f2bf(acc1[i] * inv);
        obuf[base + 32] = f2bf(acc2[i] * inv);
        obuf[base + 48] = f2bf(acc3[i] * inv);
    }
}

// ---------------------------------------------------------------------------
__global__ void concat_kernel(const float* __restrict__ x, const float* __restrict__ gt,
                              float* __restrict__ cat)
{
    const int idx = blockIdx.x * 256 + threadIdx.x;
    if (idx >= B_ * SEQ * CH) return;
    const int c = idx % CH;
    const int s = (idx / CH) % SEQ;
    const int b = idx / (CH * SEQ);
    cat[idx] = (s < NTOK) ? x[((size_t)(b * NTOK + s)) * CH + c]
                          : gt[(size_t)(s - NTOK) * CH + c];
}

__global__ void split_out_kernel(const float* __restrict__ cat, float* __restrict__ out)
{
    const int idx = blockIdx.x * 256 + threadIdx.x;
    if (idx >= B_ * SEQ * CH) return;
    const int c = idx % CH;
    const int s = (idx / CH) % SEQ;
    const int b = idx / (CH * SEQ);
    const float v = cat[idx];
    if (s < NTOK) out[((size_t)(b * NTOK + s)) * CH + c] = v;
    else out[(size_t)B_ * NTOK * CH + ((size_t)(b * GTOK + (s - NTOK))) * CH + c] = v;
}

// ---------------------------------------------------------------------------
extern "C" void kernel_launch(void* const* d_in, const int* in_sizes, int n_in,
                              void* d_out, int out_size, void* d_ws, size_t ws_size,
                              hipStream_t stream)
{
    (void)in_sizes; (void)n_in; (void)out_size; (void)ws_size;
    const float* x     = (const float*)d_in[0];
    const float* gt    = (const float*)d_in[1];
    const float* ln1g  = (const float*)d_in[2];
    const float* ln1b  = (const float*)d_in[3];
    const float* qkvw  = (const float*)d_in[4];
    const float* qkvb  = (const float*)d_in[5];
    const float* projw = (const float*)d_in[6];
    const float* projb = (const float*)d_in[7];
    const float* ln2g  = (const float*)d_in[8];
    const float* ln2b  = (const float*)d_in[9];
    const float* fc1w  = (const float*)d_in[10];
    const float* fc1b  = (const float*)d_in[11];
    const float* fc2w  = (const float*)d_in[12];
    const float* fc2b  = (const float*)d_in[13];

    // workspace layout (~202 MB)
    char* ws = (char*)d_ws;
    float* cat = (float*)ws;                    ws += (size_t)MROWS * CH * sizeof(float);
    unsigned short* act  = (unsigned short*)ws; ws += (size_t)MROWS * CH * 2;      // LN out
    unsigned short* obuf = (unsigned short*)ws; ws += (size_t)MROWS * CH * 2;      // attn out
    unsigned short* vt   = (unsigned short*)ws; ws += (size_t)MROWS * CH * 2 + 64; // V^T (+pad)
    unsigned short* big  = (unsigned short*)ws; ws += (size_t)MROWS * FF * 2;      // qkv / mlp hidden
    unsigned short* wbuf = (unsigned short*)ws;                                    // per-depth bf16 weights
    unsigned short* w_qkv  = wbuf;                                   // 3C*C
    unsigned short* w_proj = wbuf + (size_t)3 * CH * CH;             // C*C
    unsigned short* w_fc1  = wbuf + (size_t)4 * CH * CH;             // FF*C
    unsigned short* w_fc2  = wbuf + (size_t)8 * CH * CH;             // C*FF

    const int total = MROWS * CH;
    concat_kernel<<<(total + 255) / 256, 256, 0, stream>>>(x, gt, cat);

    for (int d = 0; d < DEPTH_; ++d) {
        // per-layer weight conversion fp32 -> bf16 (L2-resident afterwards)
        {
            const int n1 = 3 * CH * CH, n2 = CH * CH, n3 = FF * CH;
            f32_to_bf16_kernel<<<(n1 + 255) / 256, 256, 0, stream>>>(qkvw + (size_t)d * n1, w_qkv, n1);
            f32_to_bf16_kernel<<<(n2 + 255) / 256, 256, 0, stream>>>(projw + (size_t)d * n2, w_proj, n2);
            f32_to_bf16_kernel<<<(n3 + 255) / 256, 256, 0, stream>>>(fc1w + (size_t)d * n3, w_fc1, n3);
            f32_to_bf16_kernel<<<(n3 + 255) / 256, 256, 0, stream>>>(fc2w + (size_t)d * n3, w_fc2, n3);
        }
        // --- attention ---
        layernorm_kernel<<<MROWS, 256, 0, stream>>>(cat, ln1g + d * CH, ln1b + d * CH, act);
        {
            dim3 g(3 * CH / 64, MROWS / 128);
            gemm_wmma<<<g, 256, 0, stream>>>(act, w_qkv, qkvb + (size_t)d * 3 * CH,
                                             nullptr, nullptr, big,
                                             MROWS, 3 * CH, CH, 0);
        }
        transpose_v_kernel<<<(total + 255) / 256, 256, 0, stream>>>(big, vt);
        {
            dim3 g(SEQ / 16, HEADS_, B_);
            attn_kernel<<<g, 32, 0, stream>>>(big, vt, obuf);
        }
        {
            dim3 g(CH / 64, MROWS / 128);
            gemm_wmma<<<g, 256, 0, stream>>>(obuf, w_proj, projb + (size_t)d * CH,
                                             cat, cat, nullptr,
                                             MROWS, CH, CH, 0);
        }
        // --- MLP ---
        layernorm_kernel<<<MROWS, 256, 0, stream>>>(cat, ln2g + d * CH, ln2b + d * CH, act);
        {
            dim3 g(FF / 64, MROWS / 128);
            gemm_wmma<<<g, 256, 0, stream>>>(act, w_fc1, fc1b + (size_t)d * FF,
                                             nullptr, nullptr, big,
                                             MROWS, FF, CH, 1);
        }
        {
            dim3 g(CH / 64, MROWS / 128);
            gemm_wmma<<<g, 256, 0, stream>>>(big, w_fc2, fc2b + (size_t)d * CH,
                                             cat, cat, nullptr,
                                             MROWS, CH, FF, 0);
        }
    }

    split_out_kernel<<<(total + 255) / 256, 256, 0, stream>>>(cat, (float*)d_out);
}
